// DRL4TSP_78168404787608
// MI455X (gfx1250) — matrix operand order repaired and checked by reading, not compiled, not run on hardware
//
#include <hip/hip_runtime.h>
#include <cstdint>
#include <cstddef>

// ---------------------------------------------------------------------------
// DRL4TSP forward for MI455X (gfx1250, wave32).
//  Kernel 1: fold tiny encoder matrices, convert GEMM weights to f16.
//  Kernel 2: precompute sh / att_base / ptr_base in f16 ([B][H][SP], L2-resident).
//  Kernel 3: persistent decoder, 8 batch rows per 256-thread block, 100 steps
//            in-kernel; GRU / attention / pointer matmuls via
//            v_wmma_f32_16x16x32_f16.
// ---------------------------------------------------------------------------

typedef _Float16 half_t;
typedef __attribute__((ext_vector_type(16))) _Float16 v16h;
typedef __attribute__((ext_vector_type(8)))  float    v8f;

#define B_TOT 1024
#define SEQ   100
#define SP    112          // padded S (zero-filled tail)
#define H_DIM 128
#define NB    8            // real batch rows per block (padded to 16 for WMMA)

struct SmemDecode {
  float  gi[16][384];      // x @ W_ih^T
  float  gh[16][384];      // h @ W_hh^T
  float  ah[16][H_DIM];    // h @ Wa_h^T
  float  cp[16][H_DIM];    // context @ Wp_c^T
  float  h [NB][H_DIM];    // GRU hidden (f32)
  half_t hx[16][H_DIM];    // x (f16, rows 8..15 zero)
  half_t hh[16][H_DIM];    // h (f16, rows 8..15 zero)
  half_t hc[16][H_DIM];    // context (f16, rows 8..15 zero)
  float  dec[NB][2];       // decoder input
  float  va[H_DIM], vp[H_DIM];
  float  bih[384], bhh[384];
  float  wdec[2 * H_DIM], bdec[H_DIM];
};

// ---- WMMA fragment loaders (contiguous k = 16*(lane/16) + j in A and B, wave32) --------------------

// A: 16x32 f16 tile from LDS matrix X[16][128], K-chunk at k0.
__device__ __forceinline__ v16h load_a_frag(const half_t* X, int lane, int k0) {
  const half_t* row = X + (lane & 15) * H_DIM;
  const int kb = (lane < 16) ? 0 : 16;
  v16h a;
#pragma unroll
  for (int j = 0; j < 16; ++j) {
    a[j] = row[k0 + kb + j];
  }
  return a;
}

// B: 32x16 f16 tile, B[k][n] = W[n0+n][k0+k], W row-major [N][128] in global.
__device__ __forceinline__ v16h load_b_frag(const half_t* W, int lane, int n0, int k0) {
  const half_t* row = W + (size_t)(n0 + (lane & 15)) * H_DIM + k0 + ((lane < 16) ? 0 : 16);
  v16h b;
#pragma unroll
  for (int j = 0; j < 16; ++j) b[j] = row[j];
  return b;
}

// D(16x16) = X(16x128) * W(n0..n0+15, 0..127)^T ; write f32 tile into LDS out.
__device__ __forceinline__ void gemm_tile(const half_t* Xlds, const half_t* Wg,
                                          float* out, int ldo, int lane, int n0) {
  v8f acc = {};
#pragma unroll
  for (int kc = 0; kc < 4; ++kc) {
    v16h a = load_a_frag(Xlds, lane, kc * 32);
    v16h b = load_b_frag(Wg, lane, n0, kc * 32);
    acc = __builtin_amdgcn_wmma_f32_16x16x32_f16(false, a, false, b, (short)0, acc,
                                                 false, false);
  }
  const int n  = n0 + (lane & 15);
  const int mb = (lane < 16) ? 0 : 8;
#pragma unroll
  for (int r = 0; r < 8; ++r) out[(mb + r) * ldo + n] = acc[r];
}

__device__ __forceinline__ float wave_sum(float v) {
#pragma unroll
  for (int m = 16; m; m >>= 1) v += __shfl_xor(v, m, 32);
  return v;
}
__device__ __forceinline__ float wave_max(float v) {
#pragma unroll
  for (int m = 16; m; m >>= 1) v = fmaxf(v, __shfl_xor(v, m, 32));
  return v;
}

// ---- Kernel 1: fold small matrices + f16 weight conversion -----------------
__global__ void prep_weights(const float* __restrict__ Ws, const float* __restrict__ bs,
                             const float* __restrict__ Wd, const float* __restrict__ bd,
                             const float* __restrict__ W_ih, const float* __restrict__ W_hh,
                             const float* __restrict__ Wa, const float* __restrict__ Wp,
                             float* __restrict__ Aas, float* __restrict__ Aad,
                             float* __restrict__ Aps, float* __restrict__ batt,
                             float* __restrict__ bptr,
                             half_t* __restrict__ WihH, half_t* __restrict__ WhhH,
                             half_t* __restrict__ WahH, half_t* __restrict__ WpcH) {
  const int tid = threadIdx.x;
  if (tid < H_DIM) {
    const int h = tid;
    float as0 = 0, as1 = 0, ad0 = 0, ad1 = 0, ba = 0, ps0 = 0, ps1 = 0, bp = 0;
    for (int k = 0; k < H_DIM; ++k) {
      float was = Wa[h * 384 + k], wad = Wa[h * 384 + 128 + k], wps = Wp[h * 256 + k];
      as0 += was * Ws[k * 2];     as1 += was * Ws[k * 2 + 1];
      ad0 += wad * Wd[k * 2];     ad1 += wad * Wd[k * 2 + 1];
      ba  += was * bs[k] + wad * bd[k];
      ps0 += wps * Ws[k * 2];     ps1 += wps * Ws[k * 2 + 1];
      bp  += wps * bs[k];
    }
    Aas[2 * h] = as0; Aas[2 * h + 1] = as1;
    Aad[2 * h] = ad0; Aad[2 * h + 1] = ad1;
    Aps[2 * h] = ps0; Aps[2 * h + 1] = ps1;
    batt[h] = ba; bptr[h] = bp;
  }
  for (int i = tid; i < 384 * H_DIM; i += blockDim.x) {
    WihH[i] = (half_t)W_ih[i];
    WhhH[i] = (half_t)W_hh[i];
  }
  for (int i = tid; i < H_DIM * H_DIM; i += blockDim.x) {
    int h = i >> 7, k = i & 127;
    WahH[i] = (half_t)Wa[h * 384 + 256 + k];   // Wa_h = Wa[:, 2H:3H]
    WpcH[i] = (half_t)Wp[h * 256 + 128 + k];   // Wp_c = Wp[:, H:2H]
  }
}

// ---- Kernel 2: sh / att_base / ptr_base (f16, [B][H][SP]) ------------------
__global__ void prep_bases(const float* __restrict__ st, const float* __restrict__ dy,
                           const float* __restrict__ Ws, const float* __restrict__ bs,
                           const float* __restrict__ Aas, const float* __restrict__ Aad,
                           const float* __restrict__ Aps, const float* __restrict__ batt,
                           const float* __restrict__ bptr,
                           half_t* __restrict__ shb, half_t* __restrict__ attb,
                           half_t* __restrict__ ptb) {
  const int b = blockIdx.x, s = threadIdx.x;
  if (s >= SP) return;
  float s0 = 0, s1 = 0, d0 = 0, d1 = 0;
  if (s < SEQ) {
    s0 = st[(size_t)b * 2 * SEQ + s];       s1 = st[(size_t)b * 2 * SEQ + SEQ + s];
    d0 = dy[(size_t)b * 2 * SEQ + s];       d1 = dy[(size_t)b * 2 * SEQ + SEQ + s];
  }
  for (int h = 0; h < H_DIM; ++h) {
    float shv = 0.f, atv = 0.f, ptv = 0.f;
    if (s < SEQ) {
      shv = Ws[2 * h] * s0 + Ws[2 * h + 1] * s1 + bs[h];
      atv = Aas[2 * h] * s0 + Aas[2 * h + 1] * s1 +
            Aad[2 * h] * d0 + Aad[2 * h + 1] * d1 + batt[h];
      ptv = Aps[2 * h] * s0 + Aps[2 * h + 1] * s1 + bptr[h];
    }
    size_t idx = ((size_t)b * H_DIM + h) * SP + s;
    shb[idx] = (half_t)shv; attb[idx] = (half_t)atv; ptb[idx] = (half_t)ptv;
  }
}

// ---- Kernel 3: persistent decoder ------------------------------------------
__global__ void __launch_bounds__(256, 1)
drl4tsp_decode(const float* __restrict__ static_,
               const float* __restrict__ Wdec, const float* __restrict__ bdec,
               const float* __restrict__ b_ih, const float* __restrict__ b_hh,
               const float* __restrict__ va,   const float* __restrict__ vp,
               const half_t* __restrict__ Wih, const half_t* __restrict__ Whh,
               const half_t* __restrict__ Wah, const half_t* __restrict__ Wpc,
               const half_t* __restrict__ shb, const half_t* __restrict__ attb,
               const half_t* __restrict__ ptb, float* __restrict__ out) {
  extern __shared__ unsigned char smem_raw[];
  SmemDecode& sm = *reinterpret_cast<SmemDecode*>(smem_raw);
  const int tid = threadIdx.x, lane = tid & 31, wave = tid >> 5;
  const int b0 = blockIdx.x * NB;

  // one-time init
  for (int i = tid; i < H_DIM; i += 256) { sm.va[i] = va[i]; sm.vp[i] = vp[i]; sm.bdec[i] = bdec[i]; }
  for (int i = tid; i < 2 * H_DIM; i += 256) sm.wdec[i] = Wdec[i];
  for (int i = tid; i < 384; i += 256) { sm.bih[i] = b_ih[i]; sm.bhh[i] = b_hh[i]; }
  for (int i = tid; i < NB * H_DIM; i += 256) (&sm.h[0][0])[i] = 0.f;
  for (int i = tid; i < 16 * H_DIM; i += 256) {
    (&sm.hx[0][0])[i] = (half_t)0.f;
    (&sm.hh[0][0])[i] = (half_t)0.f;
    (&sm.hc[0][0])[i] = (half_t)0.f;
  }
  for (int i = tid; i < NB * 2; i += 256) (&sm.dec[0][0])[i] = 0.f;
  __syncthreads();

  for (int step = 0; step < SEQ; ++step) {
    // x = dec_in @ Wdec^T + bdec  (F=2, elementwise)
    for (int i = tid; i < NB * H_DIM; i += 256) {
      int b = i >> 7, k = i & 127;
      float x = sm.wdec[2 * k] * sm.dec[b][0] + sm.wdec[2 * k + 1] * sm.dec[b][1] + sm.bdec[k];
      sm.hx[b][k] = (half_t)x;
    }
    __syncthreads();

    // GRU gate matmuls: gi = x @ W_ih^T, gh = h @ W_hh^T  (WMMA, 24 N-tiles)
    for (int t = wave; t < 24; t += 8) {
      gemm_tile(&sm.hx[0][0], Wih, &sm.gi[0][0], 384, lane, t * 16);
      gemm_tile(&sm.hh[0][0], Whh, &sm.gh[0][0], 384, lane, t * 16);
    }
    __syncthreads();

    // GRU gates (torch order r,z,n)
    for (int i = tid; i < NB * H_DIM; i += 256) {
      int b = i >> 7, k = i & 127;
      float ir  = sm.gi[b][k]       + sm.bih[k]       + sm.gh[b][k]       + sm.bhh[k];
      float iz  = sm.gi[b][128 + k] + sm.bih[128 + k] + sm.gh[b][128 + k] + sm.bhh[128 + k];
      float inn = sm.gi[b][256 + k] + sm.bih[256 + k];
      float hn  = sm.gh[b][256 + k] + sm.bhh[256 + k];
      float r = 1.f / (1.f + __expf(-ir));
      float z = 1.f / (1.f + __expf(-iz));
      float n = tanhf(inn + r * hn);
      float hv = (1.f - z) * n + z * sm.h[b][k];
      sm.h[b][k]  = hv;
      sm.hh[b][k] = (half_t)hv;
    }
    __syncthreads();

    // ah = h @ Wa_h^T  (WMMA, 8 N-tiles, one per wave)
    gemm_tile(&sm.hh[0][0], Wah, &sm.ah[0][0], H_DIM, lane, wave * 16);
    __syncthreads();

    // attention scores + softmax + context  (one batch row per wave)
    {
      const int b = wave, bg = b0 + b;
      const half_t* acol = attb + (size_t)bg * H_DIM * SP;
      float sc[4];
#pragma unroll
      for (int c = 0; c < 4; ++c) {
        int s = lane + 32 * c;
        float acc = -1e30f;
        if (s < SEQ) {
          acc = 0.f;
          const half_t* p = acol + s;
          for (int k = 0; k < H_DIM; ++k)
            acc += sm.va[k] * tanhf(sm.ah[b][k] + (float)p[(size_t)k * SP]);
        }
        sc[c] = acc;
      }
      float mx = wave_max(fmaxf(fmaxf(sc[0], sc[1]), fmaxf(sc[2], sc[3])));
      float aR[4]; float ssum = 0.f;
#pragma unroll
      for (int c = 0; c < 4; ++c) {
        int s = lane + 32 * c;
        aR[c] = (s < SEQ) ? __expf(sc[c] - mx) : 0.f;
        ssum += aR[c];
      }
      float inv = 1.f / wave_sum(ssum);
#pragma unroll
      for (int c = 0; c < 4; ++c) aR[c] *= inv;
      // context[h] = sum_s a[s] * sh[b][h][s]
      const half_t* scol = shb + (size_t)bg * H_DIM * SP;
      for (int k = 0; k < H_DIM; ++k) {
        const half_t* q = scol + (size_t)k * SP + lane;
        float part = aR[0] * (float)q[0]  + aR[1] * (float)q[32] +
                     aR[2] * (float)q[64] + aR[3] * (float)q[96];
        part = wave_sum(part);
        if (lane == 0) sm.hc[b][k] = (half_t)part;
      }
    }
    __syncthreads();

    // cp = context @ Wp_c^T (WMMA)
    gemm_tile(&sm.hc[0][0], Wpc, &sm.cp[0][0], H_DIM, lane, wave * 16);
    __syncthreads();

    // pointer logits + greedy argmax + logp (one batch row per wave)
    {
      const int b = wave, bg = b0 + b;
      const half_t* pcol = ptb + (size_t)bg * H_DIM * SP;
      float sc[4];
#pragma unroll
      for (int c = 0; c < 4; ++c) {
        int s = lane + 32 * c;
        float acc = -1e30f;
        if (s < SEQ) {
          acc = 0.f;
          const half_t* p = pcol + s;
          for (int k = 0; k < H_DIM; ++k)
            acc += sm.vp[k] * tanhf(sm.cp[b][k] + (float)p[(size_t)k * SP]);
        }
        sc[c] = acc;
      }
      float bv = sc[0]; int bi = lane;  // chunk 0 always valid (lane < 32 < 100)
#pragma unroll
      for (int c = 1; c < 4; ++c) {
        int s = lane + 32 * c;
        if (sc[c] > bv) { bv = sc[c]; bi = s; }   // strict '>' keeps first occurrence
      }
#pragma unroll
      for (int m = 16; m; m >>= 1) {
        float ov = __shfl_xor(bv, m, 32);
        int   oi = __shfl_xor(bi, m, 32);
        if (ov > bv || (ov == bv && oi < bi)) { bv = ov; bi = oi; }
      }
      float se = 0.f;   // softmax(logits/TEMP): logp = -log sum exp((l - lmax)/2)
#pragma unroll
      for (int c = 0; c < 4; ++c) {
        int s = lane + 32 * c;
        if (s < SEQ) se += __expf((sc[c] - bv) * 0.5f);
      }
      se = wave_sum(se);
      if (lane == 0) {
        out[(size_t)bg * SEQ + step] = (float)bi;
        out[(size_t)B_TOT * SEQ + (size_t)bg * SEQ + step] = -__logf(se);
        sm.dec[b][0] = static_[(size_t)bg * 2 * SEQ + bi];
        sm.dec[b][1] = static_[(size_t)bg * 2 * SEQ + SEQ + bi];
      }
    }
    __syncthreads();
  }
}

// ---------------------------------------------------------------------------
extern "C" void kernel_launch(void* const* d_in, const int* in_sizes, int n_in,
                              void* d_out, int out_size, void* d_ws, size_t ws_size,
                              hipStream_t stream) {
  const float* st   = (const float*)d_in[0];
  const float* dy   = (const float*)d_in[1];
  const float* Ws   = (const float*)d_in[2];
  const float* bs   = (const float*)d_in[3];
  const float* Wd   = (const float*)d_in[4];
  const float* bd   = (const float*)d_in[5];
  const float* Wdec = (const float*)d_in[6];
  const float* bdec = (const float*)d_in[7];
  const float* W_ih = (const float*)d_in[8];
  const float* W_hh = (const float*)d_in[9];
  const float* b_ih = (const float*)d_in[10];
  const float* b_hh = (const float*)d_in[11];
  const float* va   = (const float*)d_in[12];
  const float* Wa   = (const float*)d_in[13];
  const float* vp   = (const float*)d_in[14];
  const float* Wp   = (const float*)d_in[15];
  float* out = (float*)d_out;
  (void)in_sizes; (void)n_in; (void)out_size; (void)ws_size;

  uint8_t* ws = (uint8_t*)d_ws;
  size_t off = 0;
  auto take = [&](size_t bytes) {
    uint8_t* p = ws + off;
    off = (off + bytes + 255) & ~(size_t)255;
    return p;
  };
  const size_t base_elems = (size_t)B_TOT * H_DIM * SP;
  half_t* shb  = (half_t*)take(base_elems * 2);
  half_t* attb = (half_t*)take(base_elems * 2);
  half_t* ptb  = (half_t*)take(base_elems * 2);
  half_t* WihH = (half_t*)take(384 * 128 * 2);
  half_t* WhhH = (half_t*)take(384 * 128 * 2);
  half_t* WahH = (half_t*)take(128 * 128 * 2);
  half_t* WpcH = (half_t*)take(128 * 128 * 2);
  float*  Aas  = (float*)take(256 * 4);
  float*  Aad  = (float*)take(256 * 4);
  float*  Aps  = (float*)take(256 * 4);
  float*  batt = (float*)take(128 * 4);
  float*  bptr = (float*)take(128 * 4);

  prep_weights<<<1, 256, 0, stream>>>(Ws, bs, Wd, bd, W_ih, W_hh, Wa, Wp,
                                      Aas, Aad, Aps, batt, bptr,
                                      WihH, WhhH, WahH, WpcH);
  prep_bases<<<B_TOT, 128, 0, stream>>>(st, dy, Ws, bs, Aas, Aad, Aps, batt, bptr,
                                        shb, attb, ptb);
  drl4tsp_decode<<<B_TOT / NB, 256, sizeof(SmemDecode), stream>>>(
      st, Wdec, bdec, b_ih, b_hh, va, vp, WihH, WhhH, WahH, WpcH,
      shb, attb, ptb, out);
}
